// CrissCrossAttention_11519102287929
// MI455X (gfx1250) — matrix operand order repaired and checked by reading, not compiled, not run on hardware
//
#include <hip/hip_runtime.h>
#include <hip/hip_bf16.h>

// ---------------- problem constants ----------------
constexpr int Bn = 8, C = 512, CQ = 128, H = 96, Wd = 96, HW = H * Wd;
constexpr int LDK = 40; // padded LDS row length (ushorts) to dodge bank conflicts

// ---------------- types ----------------
typedef __attribute__((ext_vector_type(16))) __bf16          v16bf;
typedef __attribute__((ext_vector_type(8)))  float           v8f;
typedef __attribute__((ext_vector_type(16))) unsigned short  v16us;
typedef __attribute__((ext_vector_type(4)))  unsigned int    u32x4;
typedef __attribute__((ext_vector_type(8)))  int             i32x8;
typedef __attribute__((ext_vector_type(4)))  int             i32x4;

// ---------------- helpers ----------------
__device__ inline unsigned short f32_to_bf16(float f) {
    unsigned int u = __builtin_bit_cast(unsigned int, f);
    u += 0x7FFFu + ((u >> 16) & 1u);            // round to nearest even
    return (unsigned short)(u >> 16);
}

// A-matrix fragment, 16x32 bf16 (ISA 7.12.2): lane<16 -> K {0..7,16..23},
// lane>=16 -> K {8..15,24..31}; LDS tile is [row][K] with row stride LDK.
__device__ inline v16bf frag_a(const unsigned short* lds, int row, int lane) {
    const unsigned short* p = lds + row * LDK + ((lane & 16) ? 8 : 0);
    v16us t;
#pragma unroll
    for (int i = 0; i < 8; ++i) { t[i] = p[i]; t[i + 8] = p[i + 16]; }
    return __builtin_bit_cast(v16bf, t);
}

// B-matrix fragment, 32x16 bf16: lane<16 -> K {0..7,16..23},
// lane>=16 -> K {8..15,24..31}; LDS tile stored K-contiguous per N row: [n][K].
__device__ inline v16bf frag_b(const unsigned short* lds, int row, int lane) {
    const unsigned short* p = lds + row * LDK + ((lane & 16) ? 8 : 0);
    v16us t;
#pragma unroll
    for (int i = 0; i < 8; ++i) { t[i] = p[i]; t[i + 8] = p[i + 16]; }
    return __builtin_bit_cast(v16bf, t);
}

__device__ inline v8f wmma_bf16(v16bf a, v16bf b, v8f c) {
    return __builtin_amdgcn_wmma_f32_16x16x32_bf16(false, a, false, b,
                                                   (short)0, c, false, false);
}

// ---- Tensor Data Mover: 2D bf16 tile (tile_d0 contiguous elems per row,
// tile_d1 rows, row stride d0_stride elems) -> LDS rows padded to LDK ushorts.
// D# per ISA ch.8: group0 {count=1, lds_addr, global_addr, type=2},
// group1 {data_size=2B, pad_enable, pad_interval=3 (16 dw = 32 u16),
//         pad_amount=3 (4 dw = 8 u16), dims/strides}.
__device__ __forceinline__ void tdm_load_2d(unsigned int lds_addr,
                                            const void* gptr,
                                            unsigned int tile_d0,
                                            unsigned int tile_d1,
                                            unsigned long long d0_stride)
{
    unsigned long long ga = (unsigned long long)gptr;
    u32x4 g0;
    g0[0] = 1u;                                        // count=1, user D#
    g0[1] = lds_addr;                                  // LDS byte offset
    g0[2] = (unsigned int)(ga & 0xFFFFFFFFu);          // global addr [31:0]
    g0[3] = (unsigned int)((ga >> 32) & 0x01FFFFFFu)   // global addr [56:32]
          | (2u << 30);                                // type = 2 (image)
    i32x8 g1;
    g1[0] = (1 << 16)      // data_size = 1 -> 2 bytes
          | (1 << 20)      // pad_enable
          | (3 << 22)      // pad_interval: 16 dwords (= 32 u16 = tile_d0)
          | (3 << 25);     // pad_amount: 4 dwords (= 8 u16 -> stride LDK)
    g1[1] = (int)((tile_d0 & 0xFFFFu) << 16);          // tensor_dim0 = tile_d0
    g1[2] = (int)((tile_d1 & 0xFFFFu) << 16);          // tensor_dim1 = tile_d1
    g1[3] = (int)(tile_d0 << 16);                      // tile_dim0
    g1[4] = (int)(tile_d1 & 0xFFFFu);                  // tile_dim1 (tile_dim2=0)
    g1[5] = (int)(d0_stride & 0xFFFFFFFFu);            // tensor_dim0_stride lo
    g1[6] = (int)((d0_stride >> 32) & 0xFFFFu);        // stride hi (dim1_stride=0)
    g1[7] = 0;
    i32x4 z4 = {0, 0, 0, 0};
#if __clang_major__ >= 23
    i32x8 z8 = {0, 0, 0, 0, 0, 0, 0, 0};
    __builtin_amdgcn_tensor_load_to_lds(g0, g1, z4, z4, z8, 0);
#else
    __builtin_amdgcn_tensor_load_to_lds(g0, g1, z4, z4, 0);
#endif
}

__device__ __forceinline__ unsigned int lds_off(const void* p) {
    // flat LDS aperture address: offset within LDS is addr[31:0]
    return (unsigned int)(unsigned long long)p;
}

// ================= kernel 1: fused qkv projection (1x1 conv as GEMM) ========
// Out[o,p] = sum_c W[o,c] * x[b,c,p] + bias[o], o in [0,768): q(128)|k(128)|v(512)
__global__ __launch_bounds__(128)
void k_proj(const float* __restrict__ x,
            const float* __restrict__ Wq, const float* __restrict__ bq,
            const float* __restrict__ Wk, const float* __restrict__ bk,
            const float* __restrict__ Wv, const float* __restrict__ bv,
            unsigned short* __restrict__ qo, unsigned short* __restrict__ ko,
            unsigned short* __restrict__ vo)
{
    __shared__ unsigned short As[64 * LDK];
    __shared__ unsigned short Bs[64 * LDK];
    const int tid = threadIdx.x, lane = tid & 31, wave = tid >> 5;
    const int p0 = blockIdx.x * 64;     // pixel block
    const int m0 = blockIdx.y * 64;     // fused output-channel block
    const int b  = blockIdx.z;

    const float* Wsel; const float* bsel; unsigned short* dst; int off, Cd;
    if (m0 < 128)      { Wsel = Wq; bsel = bq; dst = qo; off = 0;   Cd = CQ; }
    else if (m0 < 256) { Wsel = Wk; bsel = bk; dst = ko; off = 128; Cd = CQ; }
    else               { Wsel = Wv; bsel = bv; dst = vo; off = 256; Cd = C;  }
    const int mw = m0 - off;

    v8f acc[4] = {};
    for (int k0 = 0; k0 < C; k0 += 32) {
        // prefetch next K-step of x into L2 (global_prefetch_b8)
        if (k0 + 32 < C)
            __builtin_prefetch(&x[(b * C + k0 + 32 + (tid >> 6)) * HW + p0 + (tid & 63)], 0, 1);
        // stage A: 64x32 weights -> bf16
#pragma unroll
        for (int i = 0; i < 16; ++i) {
            int idx = tid + i * 128;
            int c = idx & 31, r = idx >> 5;
            As[r * LDK + c] = f32_to_bf16(Wsel[(mw + r) * C + k0 + c]);
        }
        // stage B: 32(K) x 64(p) of x, stored K-contiguous per pixel row
#pragma unroll
        for (int i = 0; i < 16; ++i) {
            int idx = tid + i * 128;
            int p = idx & 63, kc = idx >> 6;
            Bs[p * LDK + kc] = f32_to_bf16(x[(b * C + k0 + kc) * HW + p0 + p]);
        }
        __syncthreads();
        v16bf a = frag_a(As, wave * 16 + (lane & 15), lane);
#pragma unroll
        for (int nb = 0; nb < 4; ++nb) {
            v16bf bfr = frag_b(Bs, nb * 16 + (lane & 15), lane);
            acc[nb] = wmma_bf16(a, bfr, acc[nb]);
        }
        __syncthreads();
    }
    const int r0 = wave * 16 + ((lane & 16) ? 8 : 0);
    const int nc = lane & 15;
#pragma unroll
    for (int nb = 0; nb < 4; ++nb) {
        int p = p0 + nb * 16 + nc;
#pragma unroll
        for (int r = 0; r < 8; ++r) {
            int row = mw + r0 + r;
            dst[(b * Cd + row) * HW + p] = f32_to_bf16(acc[nb][r] + bsel[row]);
        }
    }
}

// ================= kernel 2: energies (96x96x128 GEMM per slice) ============
// mode 0 (y=0): energy_W[b,h=s][w,j] = sum_c q[b,c,s,w]*k[b,c,s,j] -> E[...,96+j]
// mode 1 (y=1): energy_H[b,w=s][h,j] = sum_c q[b,c,h,s]*k[b,c,j,s] -> E[...,j], diag=-inf
__global__ __launch_bounds__(192)
void k_energy(const unsigned short* __restrict__ q,
              const unsigned short* __restrict__ k,
              float* __restrict__ E)
{
    __shared__ unsigned short As[96 * LDK];
    __shared__ unsigned short Bs[96 * LDK];
    const int tid = threadIdx.x, lane = tid & 31, wave = tid >> 5;
    const int b = blockIdx.x / 96, s = blockIdx.x % 96;
    const int mode = blockIdx.y;
    const int u = tid % 96, half = tid / 96;

    v8f acc[6] = {};
    for (int k0 = 0; k0 < CQ; k0 += 32) {
#pragma unroll
        for (int i = 0; i < 16; ++i) {
            int c = i * 2 + half;
            int gi = (mode == 0) ? (b * CQ + k0 + c) * HW + s * Wd + u
                                 : (b * CQ + k0 + c) * HW + u * Wd + s;
            As[u * LDK + c] = q[gi];
            Bs[u * LDK + c] = k[gi];
        }
        __syncthreads();
        v16bf a = frag_a(As, wave * 16 + (lane & 15), lane);
#pragma unroll
        for (int nb = 0; nb < 6; ++nb) {
            v16bf bfr = frag_b(Bs, nb * 16 + (lane & 15), lane);
            acc[nb] = wmma_bf16(a, bfr, acc[nb]);
        }
        __syncthreads();
    }
    const int r0 = wave * 16 + ((lane & 16) ? 8 : 0);
    const int nc0 = lane & 15;
#pragma unroll
    for (int nb = 0; nb < 6; ++nb)
#pragma unroll
        for (int r = 0; r < 8; ++r) {
            int mr = r0 + r, jc = nb * 16 + nc0;
            float vv = acc[nb][r];
            if (mode == 0)
                E[((b * H + s) * Wd + mr) * 192 + 96 + jc] = vv;
            else
                E[((b * H + mr) * Wd + s) * 192 + jc] =
                    (mr == jc) ? -__builtin_inff() : vv;
        }
}

// ================= kernel 3: softmax over 192, bf16 in place ================
__global__ __launch_bounds__(256)
void k_softmax(float* __restrict__ E)
{
    const int lane = threadIdx.x & 31;
    const int row  = (blockIdx.x * blockDim.x + threadIdx.x) >> 5;
    float* r = E + (size_t)row * 192;
    float vals[6];
    float m = -__builtin_inff();
#pragma unroll
    for (int i = 0; i < 6; ++i) { vals[i] = r[lane + i * 32]; m = fmaxf(m, vals[i]); }
#pragma unroll
    for (int off = 16; off; off >>= 1) m = fmaxf(m, __shfl_xor(m, off, 32));
    float s = 0.f;
#pragma unroll
    for (int i = 0; i < 6; ++i) { vals[i] = __expf(vals[i] - m); s += vals[i]; }
#pragma unroll
    for (int off = 16; off; off >>= 1) s += __shfl_xor(s, off, 32);
    float inv = 1.f / s;
    unsigned short* o = (unsigned short*)r;
#pragma unroll
    for (int i = 0; i < 6; ++i) o[lane + i * 32] = f32_to_bf16(vals[i] * inv);
}

// ================= kernel 4a: out_H = V * att_H^T per (b,w) =================
// B (attention) tile staged by the Tensor Data Mover; A tile is a doubly
// strided gather (kept as manual loads, overlapped with the DMA).
__global__ __launch_bounds__(128)
void k_outH(const unsigned short* __restrict__ v,
            const unsigned short* __restrict__ att,
            float* __restrict__ out)
{
    __shared__ unsigned short As[64 * LDK];
    __shared__ unsigned short Bs[96 * LDK];
    const int tid = threadIdx.x, lane = tid & 31, wave = tid >> 5;
    const int b = blockIdx.x / 96, w = blockIdx.x % 96;
    const int c0 = blockIdx.y * 64;

    v8f acc[6] = {};
    for (int k0 = 0; k0 < H; k0 += 32) {
        if (wave == 0)  // one TDM descriptor: 96 rows x 32 u16, row stride 384
            tdm_load_2d(lds_off(Bs), &att[((size_t)(b * H) * Wd + w) * 384 + k0],
                        32, 96, 384ull);
#pragma unroll
        for (int i = 0; i < 16; ++i) {
            int idx = tid + i * 128;
            int kj = idx & 31, m = idx >> 5;
            As[m * LDK + kj] = v[(b * C + c0 + m) * HW + (k0 + kj) * Wd + w];
        }
        if (wave == 0) __builtin_amdgcn_s_wait_tensorcnt(0);
        __syncthreads();
        v16bf a = frag_a(As, wave * 16 + (lane & 15), lane);
#pragma unroll
        for (int nb = 0; nb < 6; ++nb) {
            v16bf bfr = frag_b(Bs, nb * 16 + (lane & 15), lane);
            acc[nb] = wmma_bf16(a, bfr, acc[nb]);
        }
        __syncthreads();
    }
    const int r0 = wave * 16 + ((lane & 16) ? 8 : 0);
    const int nc0 = lane & 15;
#pragma unroll
    for (int nb = 0; nb < 6; ++nb)
#pragma unroll
        for (int r = 0; r < 8; ++r) {
            int cc = c0 + r0 + r, h = nb * 16 + nc0;
            out[(b * C + cc) * HW + h * Wd + w] = acc[nb][r];
        }
}

// ====== kernel 4b: out_W per (b,h), fuse gamma*(outH+outW)+x residual =======
// Both A (v rows) and B (attention rows) tiles staged by the Tensor Data Mover.
__global__ __launch_bounds__(128)
void k_outW(const unsigned short* __restrict__ v,
            const unsigned short* __restrict__ att,
            const float* __restrict__ x,
            const float* __restrict__ gamma,
            float* __restrict__ out)
{
    __shared__ unsigned short As[64 * LDK];
    __shared__ unsigned short Bs[96 * LDK];
    const int tid = threadIdx.x, lane = tid & 31, wave = tid >> 5;
    const int b = blockIdx.x / 96, h = blockIdx.x % 96;
    const int c0 = blockIdx.y * 64;

    v8f acc[6] = {};
    for (int k0 = 0; k0 < Wd; k0 += 32) {
        if (wave == 0) {
            // A: 64 rows (channels) x 32 u16, row stride HW
            tdm_load_2d(lds_off(As),
                        &v[(size_t)(b * C + c0) * HW + h * Wd + k0],
                        32, 64, (unsigned long long)HW);
            // B: 96 rows (w) x 32 u16, row stride 384
            tdm_load_2d(lds_off(Bs),
                        &att[((size_t)(b * H + h) * Wd) * 384 + 96 + k0],
                        32, 96, 384ull);
            __builtin_amdgcn_s_wait_tensorcnt(0);
        }
        __syncthreads();
        v16bf a = frag_a(As, wave * 16 + (lane & 15), lane);
#pragma unroll
        for (int nb = 0; nb < 6; ++nb) {
            v16bf bfr = frag_b(Bs, nb * 16 + (lane & 15), lane);
            acc[nb] = wmma_bf16(a, bfr, acc[nb]);
        }
        __syncthreads();
    }
    const float g = gamma[0];
    const int r0 = wave * 16 + ((lane & 16) ? 8 : 0);
    const int nc0 = lane & 15;
#pragma unroll
    for (int nb = 0; nb < 6; ++nb)
#pragma unroll
        for (int r = 0; r < 8; ++r) {
            int cc = c0 + r0 + r, wc = nb * 16 + nc0;
            int idx = (b * C + cc) * HW + h * Wd + wc;
            out[idx] = g * (out[idx] + acc[nb][r]) + x[idx];
        }
}

// ================================ launcher ==================================
extern "C" void kernel_launch(void* const* d_in, const int* in_sizes, int n_in,
                              void* d_out, int out_size, void* d_ws, size_t ws_size,
                              hipStream_t stream)
{
    const float* x  = (const float*)d_in[0];
    const float* Wq = (const float*)d_in[1];
    const float* bq = (const float*)d_in[2];
    const float* Wk = (const float*)d_in[3];
    const float* bk = (const float*)d_in[4];
    const float* Wv = (const float*)d_in[5];
    const float* bv = (const float*)d_in[6];
    const float* gm = (const float*)d_in[7];
    float* out = (float*)d_out;

    char* ws = (char*)d_ws;
    unsigned short* qo  = (unsigned short*)(ws);                 // 18.9 MB
    unsigned short* ko  = (unsigned short*)(ws + 18874368);      // 18.9 MB
    unsigned short* vo  = (unsigned short*)(ws + 37748736);      // 75.5 MB
    float*          E   = (float*)(ws + 113246208);              // 56.6 MB
    unsigned short* att = (unsigned short*)E;                    // reused in place

    k_proj   <<<dim3(HW / 64, 768 / 64, Bn), 128, 0, stream>>>(x, Wq, bq, Wk, bk, Wv, bv, qo, ko, vo);
    k_energy <<<dim3(Bn * 96, 2),            192, 0, stream>>>(qo, ko, E);
    k_softmax<<<dim3(Bn * HW / 8),           256, 0, stream>>>(E);
    k_outH   <<<dim3(Bn * 96, C / 64),       128, 0, stream>>>(vo, att, out);
    k_outW   <<<dim3(Bn * 96, C / 64),       128, 0, stream>>>(vo, att, x, gm, out);
}
